// BaseAttention_25125558681894
// MI455X (gfx1250) — matrix operand order repaired and checked by reading, not compiled, not run on hardware
//
#include <hip/hip_runtime.h>

// Problem constants
#define NB 8        // batch
#define NT 1024     // sequence length
#define DM 512      // d_model
#define KD 64       // key/value head dim
#define NSP 4       // s-loop split factor (flash-decode style partials)

typedef __bf16 bf16;
typedef bf16  v16bf __attribute__((ext_vector_type(16)));
typedef bf16  v8bf  __attribute__((ext_vector_type(8)));
typedef bf16  v2bf  __attribute__((ext_vector_type(2)));
typedef float v8f   __attribute__((ext_vector_type(8)));
typedef float v8ff  __attribute__((ext_vector_type(8)));
typedef float v2f   __attribute__((ext_vector_type(2)));

union FragA { v16bf v; v8bf half8[2]; unsigned short h[16]; }; // 16x32 bf16 A
union FragB { v16bf v; unsigned u[8]; };                       // 32x16 bf16 B

__device__ __forceinline__ v8f wmma_bf16(v16bf a, v16bf b, v8f c) {
  // D(16x16,f32) = A(16x32,bf16) * B(32x16,bf16) + C
  return __builtin_amdgcn_wmma_f32_16x16x32_bf16(false, a, false, b, (short)0, c,
                                                 false, false);
}

// packed f32x2 -> bf16x2 (lowers to v_cvt_pk_bf16_f32)
__device__ __forceinline__ unsigned pack2bf(float a, float b) {
  v2f f = {a, b};
  union { v2bf v; unsigned u; } x;
  x.v = __builtin_convertvector(f, v2bf);
  return x.u;
}

// packed conversion of 8 consecutive (32B-aligned) floats -> bf16x8
__device__ __forceinline__ v8bf cvt8(const float* p) {
  return __builtin_convertvector(*(const v8ff*)p, v8bf);
}

// ---------------------------------------------------------------------------
// Projection: Y(bf16, M x 64) = X(f32, M x 512) @ W(f32, 512 x 64)
// ---------------------------------------------------------------------------
__global__ __launch_bounds__(32)
void proj16x64(const float* __restrict__ X, const float* __restrict__ W,
               bf16* __restrict__ Y) {
  const int m0   = blockIdx.x * 16;
  const int lane = threadIdx.x;
  const int l15  = lane & 15;
  const bool hi  = (lane & 16) != 0;

  v8f c[4];
  #pragma unroll
  for (int nt = 0; nt < 4; ++nt) c[nt] = {};

  const float* xrow = X + (size_t)(m0 + l15) * DM;

  for (int k0 = 0; k0 < DM; k0 += 32) {
    FragA a;
    const int ka = k0 + (hi ? 16 : 0);
    a.half8[0] = cvt8(xrow + ka);
    a.half8[1] = cvt8(xrow + ka + 8);
    #pragma unroll
    for (int nt = 0; nt < 4; ++nt) {
      FragB b;
      #pragma unroll
      for (int j = 0; j < 8; ++j) {
        const int k = k0 + 2 * j + (hi ? 16 : 0);
        const float* wp = W + (size_t)k * KD + nt * 16 + l15;
        b.u[j] = pack2bf(wp[0], wp[KD]);
      }
      c[nt] = wmma_bf16(a.v, b.v, c[nt]);
    }
  }
  #pragma unroll
  for (int nt = 0; nt < 4; ++nt) {
    #pragma unroll
    for (int r = 0; r < 8; ++r) {
      const int m = m0 + r + (hi ? 8 : 0);
      Y[(size_t)m * KD + nt * 16 + l15] = (bf16)c[nt][r];
    }
  }
}

// ---------------------------------------------------------------------------
// Fused attention partial. grid = (T/16 query tiles, NSP s-strides).
// Block (8 waves) covers all 8 batches; each emb tile is read exactly once
// across the whole grid. Emits unnormalized (o+rel, m, l) partials.
// ---------------------------------------------------------------------------
__global__ __launch_bounds__(256)
void attn_partial(const bf16* __restrict__ qWh, const bf16* __restrict__ kWh,
                  const bf16* __restrict__ vWh, const float* __restrict__ alpha,
                  const float* __restrict__ embQ, const float* __restrict__ embS,
                  float* __restrict__ part_o, float* __restrict__ part_ml) {
  const int t0   = blockIdx.x * 16;
  const int sp   = blockIdx.y;
  const int tid  = threadIdx.x;
  const int w    = tid >> 5;          // wave id == batch id for QK/PV
  const int lane = tid & 31;
  const int l15  = lane & 15;
  const bool hi  = (lane & 16) != 0;
  const int mb   = hi ? 8 : 0;

  __shared__ alignas(16) float sc[NB][16][32];   // raw scores
  __shared__ alignas(16) bf16  pz[NB][16][32];   // softmax p
  __shared__ float ff[16][16];                   // rescale factors
  __shared__ alignas(16) float outb[NB][16][KD]; // partial staging

  // QK A-fragments: qW rows t0..t0+15 of batch w
  FragA aq[2];
  {
    const bf16* qrow = qWh + ((size_t)w * NT + t0 + l15) * KD;
    #pragma unroll
    for (int kh = 0; kh < 2; ++kh) {
      const int base = kh * 32 + (hi ? 16 : 0);
      aq[kh].half8[0] = *(const v8bf*)(qrow + base);
      aq[kh].half8[1] = *(const v8bf*)(qrow + base + 8);
    }
  }
  // rel_q B-fragments: B[k][n=b] = qW[b][tau][k]
  const int tau0 = 2 * w, tau1 = 2 * w + 1;
  FragB bq[2][2];
  #pragma unroll
  for (int ti = 0; ti < 2; ++ti) {
    const int tg = t0 + (ti ? tau1 : tau0);
    #pragma unroll
    for (int kh = 0; kh < 2; ++kh) {
      #pragma unroll
      for (int j = 0; j < 8; ++j) {
        const int k = kh * 32 + 2 * j + (hi ? 16 : 0);
        unsigned u = 0u;
        if (l15 < NB)
          u = *(const unsigned*)(qWh + ((size_t)l15 * NT + tg) * KD + k);
        bq[ti][kh].u[j] = u;
      }
    }
  }

  v8f o[4];      // PV accum (batch w)
  v8f ra[2][4];  // rel_s accum (rows = batch, per tau)
  #pragma unroll
  for (int nt = 0; nt < 4; ++nt) { o[nt] = {}; ra[0][nt] = {}; ra[1][nt] = {}; }

  const int rb = tid >> 4;   // row-thread: batch (0..7 valid)
  const int rt = tid & 15;   // row-thread: t within tile
  float coef = 0.f;
  if (rb < NB) coef = 0.125f * alpha[t0 + rt];
  float m_run = -3.0e38f, l_run = 0.f;

  for (int s0 = 32 * sp; s0 < t0 + 16; s0 += 32 * NSP) {
    // (1) QK^T for batch w
    #pragma unroll
    for (int nt = 0; nt < 2; ++nt) {
      v8f d = {};
      const int srow = s0 + nt * 16 + l15;
      const bf16* kr = kWh + ((size_t)w * NT + srow) * KD;
      #pragma unroll
      for (int kh = 0; kh < 2; ++kh) {
        FragB bk;
        #pragma unroll
        for (int j = 0; j < 8; ++j)
          bk.u[j] = *(const unsigned*)(kr + kh * 32 + 2 * j + (hi ? 16 : 0));
        d = wmma_bf16(aq[kh].v, bk.v, d);
      }
      const int col = nt * 16 + l15;
      #pragma unroll
      for (int r = 0; r < 8; ++r) sc[w][mb + r][col] = d[r];
    }
    __syncthreads();

    // (2) rel_q (swapped: D[s',b] = emb_Q[tau] * qW^T) added into sc
    #pragma unroll
    for (int ti = 0; ti < 2; ++ti) {
      const int tl = ti ? tau1 : tau0;
      const float* eQ = embQ + (size_t)(t0 + tl) * NT * KD;
      #pragma unroll
      for (int nt = 0; nt < 2; ++nt) {
        v8f d = {};
        const int srow = s0 + nt * 16 + l15;
        const float* er = eQ + (size_t)srow * KD;
        #pragma unroll
        for (int kh = 0; kh < 2; ++kh) {
          FragA ae;
          const int base = kh * 32 + (hi ? 16 : 0);
          ae.half8[0] = cvt8(er + base);
          ae.half8[1] = cvt8(er + base + 8);
          d = wmma_bf16(ae.v, bq[ti][kh].v, d);
        }
        if (l15 < NB) {
          #pragma unroll
          for (int r = 0; r < 8; ++r)
            sc[l15][tl][nt * 16 + mb + r] += d[r];
        }
      }
    }
    __syncthreads();

    // (3) scale, causal mask, online softmax per row
    if (rb < NB) {
      float v[32];
      float mt = -3.0e38f;
      #pragma unroll
      for (int j = 0; j < 32; ++j) {
        float x = sc[rb][rt][j] * coef;
        if (s0 + j > t0 + rt) x += -1.0e9f;
        v[j] = x;
        mt = fmaxf(mt, x);
      }
      const float m_new = fmaxf(m_run, mt);
      const float f = __expf(m_run - m_new);
      float sum = 0.f;
      #pragma unroll
      for (int j = 0; j < 32; ++j) {
        const float p = __expf(v[j] - m_new);
        pz[rb][rt][j] = (bf16)p;
        sum += p;
      }
      l_run = l_run * f + sum;
      m_run = m_new;
      ff[rb][rt] = f;
    } else {
      ff[rb][rt] = 0.f;
    }
    __syncthreads();

    // (4) rescale accumulators, then PV and rel_s
    #pragma unroll
    for (int r = 0; r < 8; ++r) {
      const float fo = ff[w][mb + r];
      #pragma unroll
      for (int nt = 0; nt < 4; ++nt) o[nt][r] *= fo;
    }
    #pragma unroll
    for (int ti = 0; ti < 2; ++ti) {
      const int tl = ti ? tau1 : tau0;
      #pragma unroll
      for (int r = 0; r < 8; ++r) {
        const float fr = ff[mb + r][tl];
        #pragma unroll
        for (int nt = 0; nt < 4; ++nt) ra[ti][nt][r] *= fr;
      }
    }

    { // PV: A = p tile of batch w, B = vW rows s0..s0+31
      FragA ap;
      const bf16* pr = &pz[w][l15][0];
      const int base = hi ? 16 : 0;
      ap.half8[0] = *(const v8bf*)(pr + base);
      ap.half8[1] = *(const v8bf*)(pr + base + 8);
      #pragma unroll
      for (int nt = 0; nt < 4; ++nt) {
        FragB bv;
        #pragma unroll
        for (int j = 0; j < 8; ++j) {
          const int k = 2 * j + (hi ? 16 : 0);
          const bf16* vr = vWh + ((size_t)w * NT + s0 + k) * KD + nt * 16 + l15;
          union { bf16 h[2]; unsigned u; } t2;
          t2.h[0] = vr[0]; t2.h[1] = vr[KD];
          bv.u[j] = t2.u;
        }
        o[nt] = wmma_bf16(ap.v, bv.v, o[nt]);
      }
    }

    // rel_s: A = p[:, tau, s-tile] (batch as M, padded), B = emb_S[tau]
    #pragma unroll
    for (int ti = 0; ti < 2; ++ti) {
      const int tl = ti ? tau1 : tau0;
      FragA app;
      if (l15 < NB) {
        const bf16* pr = &pz[l15][tl][0];
        const int base = hi ? 16 : 0;
        app.half8[0] = *(const v8bf*)(pr + base);
        app.half8[1] = *(const v8bf*)(pr + base + 8);
      } else {
        #pragma unroll
        for (int i = 0; i < 16; ++i) app.v[i] = (bf16)0.f;
      }
      const float* eS = embS + ((size_t)(t0 + tl) * NT + s0) * KD;
      #pragma unroll
      for (int nt = 0; nt < 4; ++nt) {
        FragB bs;
        #pragma unroll
        for (int j = 0; j < 8; ++j) {
          const int k = 2 * j + (hi ? 16 : 0);
          const float* ep = eS + (size_t)k * KD + nt * 16 + l15;
          bs.u[j] = pack2bf(ep[0], ep[KD]);
        }
        ra[ti][nt] = wmma_bf16(app.v, bs.v, ra[ti][nt]);
      }
    }
  }

  // ---- stage PV accum, add rel_s, emit unnormalized partial ---------------
  #pragma unroll
  for (int nt = 0; nt < 4; ++nt) {
    #pragma unroll
    for (int r = 0; r < 8; ++r)
      outb[w][mb + r][nt * 16 + l15] = o[nt][r];
  }
  __syncthreads();

  if (!hi) {   // lanes 0..15: ra rows M = r = batch 0..7 (valid half)
    #pragma unroll
    for (int ti = 0; ti < 2; ++ti) {
      const int tl = ti ? tau1 : tau0;
      #pragma unroll
      for (int nt = 0; nt < 4; ++nt) {
        #pragma unroll
        for (int r = 0; r < 8; ++r)
          outb[r][tl][nt * 16 + l15] += ra[ti][nt][r];
      }
    }
  }
  __syncthreads();

  float* po = part_o + ((size_t)(blockIdx.x * NSP + sp)) * (NB * 16 * KD);
  for (int j = tid; j < NB * 16 * KD; j += 256)
    po[j] = (&outb[0][0][0])[j];
  if (rb < NB) {
    float* pm = part_ml + ((size_t)(blockIdx.x * NSP + sp) * 128 + tid) * 2;
    pm[0] = m_run;
    pm[1] = l_run;
  }
}

// ---------------------------------------------------------------------------
// Combine NSP partials per query tile: out = sum_i o_i*exp(m_i-m) / sum_i l_i*exp(m_i-m)
// ---------------------------------------------------------------------------
__global__ __launch_bounds__(256)
void attn_combine(const float* __restrict__ part_o, const float* __restrict__ part_ml,
                  float* __restrict__ out) {
  const int tt  = blockIdx.x;   // query tile
  const int tid = threadIdx.x;
  __shared__ float wf[NSP][NB][16];
  __shared__ float linv[NB][16];

  if (tid < 128) {
    const int b = tid >> 4, t = tid & 15;
    float m[NSP], l[NSP];
    #pragma unroll
    for (int sp = 0; sp < NSP; ++sp) {
      const float* pm = part_ml + ((size_t)(tt * NSP + sp) * 128 + tid) * 2;
      m[sp] = pm[0]; l[sp] = pm[1];
    }
    float mg = m[0];
    #pragma unroll
    for (int sp = 1; sp < NSP; ++sp) mg = fmaxf(mg, m[sp]);
    float lsum = 0.f;
    #pragma unroll
    for (int sp = 0; sp < NSP; ++sp) {
      const float wv = __expf(m[sp] - mg);   // 0 for empty partials (m=-3e38)
      wf[sp][b][t] = wv;
      lsum += l[sp] * wv;
    }
    linv[b][t] = 1.f / lsum;                 // sp=0 always non-empty -> lsum > 0
  }
  __syncthreads();

  for (int j = tid; j < NB * 16 * KD; j += 256) {
    const int b = j >> 10, t = (j >> 6) & 15, v = j & 63;
    float acc = 0.f;
    #pragma unroll
    for (int sp = 0; sp < NSP; ++sp)
      acc += part_o[((size_t)(tt * NSP + sp)) * (NB * 16 * KD) + j] * wf[sp][b][t];
    out[((size_t)b * NT + tt * 16 + t) * KD + v] = acc * linv[b][t];
  }
}

// ---------------------------------------------------------------------------
extern "C" void kernel_launch(void* const* d_in, const int* in_sizes, int n_in,
                              void* d_out, int out_size, void* d_ws, size_t ws_size,
                              hipStream_t stream) {
  (void)in_sizes; (void)n_in; (void)out_size; (void)ws_size;
  const float* query = (const float*)d_in[0];
  const float* value = (const float*)d_in[1];   // dict order: q, v, k
  const float* key   = (const float*)d_in[2];
  const float* W_Q   = (const float*)d_in[3];
  const float* W_K   = (const float*)d_in[4];
  const float* W_V   = (const float*)d_in[5];
  const float* alpha = (const float*)d_in[6];
  const float* embQ  = (const float*)d_in[7];
  const float* embS  = (const float*)d_in[8];
  float* out = (float*)d_out;

  // workspace layout
  bf16* qWh = (bf16*)d_ws;                               // 1 MB
  bf16* kWh = qWh + (size_t)NB * NT * KD;                // 1 MB
  bf16* vWh = kWh + (size_t)NB * NT * KD;                // 1 MB
  float* part_o  = (float*)(vWh + (size_t)NB * NT * KD); // 64*NSP*8192 f32 = 8 MB
  float* part_ml = part_o + (size_t)(NT / 16) * NSP * NB * 16 * KD; // 256 KB

  const dim3 pgrid(NB * NT / 16);
  proj16x64<<<pgrid, 32, 0, stream>>>(query, W_Q, qWh);
  proj16x64<<<pgrid, 32, 0, stream>>>(key,   W_K, kWh);
  proj16x64<<<pgrid, 32, 0, stream>>>(value, W_V, vWh);

  attn_partial<<<dim3(NT / 16, NSP), 256, 0, stream>>>(qWh, kWh, vWh, alpha,
                                                       embQ, embS, part_o, part_ml);
  attn_combine<<<dim3(NT / 16), 256, 0, stream>>>(part_o, part_ml, out);
}
